// MultiHeadAttention_58823872086273
// MI455X (gfx1250) — matrix operand order repaired and checked by reading, not compiled, not run on hardware
//
#include <hip/hip_runtime.h>
#include <hip/hip_bf16.h>

// ---------------------------------------------------------------------------
// MHA with relative-position bias for MI455X (gfx1250), wave32 + WMMA bf16.
// N=4, S=1024, H=1024, NH=16, D=64, P=20.
// All WMMA fragments are fed by 128-bit vector loads (bf16 pre-packed,
// weights pre-transposed, V stored d-major); predictable streams are
// prefetched ahead with unconditional (speculative) global_prefetch_b8.
// ---------------------------------------------------------------------------

#define N_B   4
#define S_LEN 1024
#define H_DIM 1024
#define NH_   16
#define D_    64
#define P_    20
#define NREL  41          // 2P+1
#define QRS   48          // padded Qrel stride

typedef __attribute__((ext_vector_type(16))) __bf16 v16bf;
typedef __attribute__((ext_vector_type(8)))  float  v8f;

union BF16Frag {
    v16bf v;
    unsigned short u[16];
    uint4 q[2];            // two 128-bit halves
};

__device__ __forceinline__ unsigned short f2bf(float f) {
    unsigned int x = __float_as_uint(f);
    return (unsigned short)((x + 0x7fffu + ((x >> 16) & 1u)) >> 16);   // RNE
}
__device__ __forceinline__ float bf2f(unsigned short u) {
    return __uint_as_float(((unsigned int)u) << 16);
}
__device__ __forceinline__ v8f v8f_zero() {
    v8f z;
    #pragma unroll
    for (int i = 0; i < 8; ++i) z[i] = 0.0f;
    return z;
}

// ---------------------------------------------------------------------------
// Pack kernels: f32 -> bf16 (straight), and f32 -> bf16 transposed (weights).
// ---------------------------------------------------------------------------
__global__ void pack_bf16_kernel(const float* __restrict__ src,
                                 unsigned short* __restrict__ dst, int n4)
{
    const int i = blockIdx.x * blockDim.x + threadIdx.x;
    if (i >= n4) return;
    const float4 f = ((const float4*)src)[i];
    ushort4 o;
    o.x = f2bf(f.x); o.y = f2bf(f.y); o.z = f2bf(f.z); o.w = f2bf(f.w);
    ((ushort4*)dst)[i] = o;
}

// Wt[n][k] = W[k][n]  (H x H)
__global__ void pack_wT_kernel(const float* __restrict__ W,
                               unsigned short* __restrict__ Wt)
{
    const int idx = blockIdx.x * blockDim.x + threadIdx.x;   // H*H threads
    const int n = idx & (H_DIM - 1);
    const int k = idx >> 10;
    Wt[(size_t)n * H_DIM + k] = f2bf(W[(size_t)k * H_DIM + n]);
}

// ---------------------------------------------------------------------------
// Kernel 1: QKV projections.  C = X @ W + b.
// A: bf16 row-major (M x K).  B: bf16 transposed (N x K).  All frag loads b128.
// Q,K outputs head-major (b,h,s,d); V output d-major (b,h,d,s).
// grid = (M/64, H/64, 3), block = 128 (4 waves, each 16 rows x 64 cols)
// ---------------------------------------------------------------------------
__global__ __launch_bounds__(128) void qkv_proj_kernel(
    const unsigned short* __restrict__ Xqb, const unsigned short* __restrict__ Xkb,
    const unsigned short* __restrict__ Xvb,
    const unsigned short* __restrict__ Wqt, const unsigned short* __restrict__ Wkt,
    const unsigned short* __restrict__ Wvt,
    const float* __restrict__ bq, const float* __restrict__ bk,
    const float* __restrict__ bv,
    unsigned short* __restrict__ Qh, unsigned short* __restrict__ Kh,
    unsigned short* __restrict__ Vt)
{
    const int z = blockIdx.z;
    const unsigned short* A  = (z == 0) ? Xqb : (z == 1) ? Xkb : Xvb;
    const unsigned short* Wt = (z == 0) ? Wqt : (z == 1) ? Wkt : Wvt;
    const float* bias        = (z == 0) ? bq  : (z == 1) ? bk  : bv;
    unsigned short* dst      = (z == 0) ? Qh  : (z == 1) ? Kh  : Vt;

    const int lane  = threadIdx.x & 31;
    const int w     = threadIdx.x >> 5;
    const int ln    = lane & 15;
    const int hi    = (lane & 16) ? 8 : 0;
    const int koffA = (lane & 16) ? 16 : 0;    // A-frag K offset per half
    const int koffB = (lane & 16) ? 16 : 0;    // B-frag K offset per half
    const int m0    = blockIdx.x * 64 + w * 16;
    const int n0    = blockIdx.y * 64;

    v8f acc[4];
    #pragma unroll
    for (int i = 0; i < 4; ++i) acc[i] = v8f_zero();

    const unsigned short* Arow = A + (size_t)(m0 + ln) * H_DIM;
    const unsigned short* Brow[4];
    #pragma unroll
    for (int nb = 0; nb < 4; ++nb)
        Brow[nb] = Wt + (size_t)(n0 + nb * 16 + ln) * H_DIM;

    for (int kt = 0; kt < H_DIM / 32; ++kt) {
        const int k0 = kt * 32;
        // speculative prefetch two k-tiles ahead (global_prefetch_b8, no guard:
        // over-the-end targets stay inside the packed workspace allocation)
        const int kpf = k0 + 64;
        __builtin_prefetch(Arow + kpf, 0, 3);
        __builtin_prefetch(Brow[0] + kpf, 0, 3);
        __builtin_prefetch(Brow[1] + kpf, 0, 3);
        __builtin_prefetch(Brow[2] + kpf, 0, 3);
        __builtin_prefetch(Brow[3] + kpf, 0, 3);

        BF16Frag a;
        a.q[0] = *(const uint4*)(Arow + k0 + koffA);
        a.q[1] = *(const uint4*)(Arow + k0 + koffA + 8);
        #pragma unroll
        for (int nb = 0; nb < 4; ++nb) {
            BF16Frag bw;
            bw.q[0] = *(const uint4*)(Brow[nb] + k0 + koffB);
            bw.q[1] = *(const uint4*)(Brow[nb] + k0 + koffB + 8);
            acc[nb] = __builtin_amdgcn_wmma_f32_16x16x32_bf16(
                false, a.v, false, bw.v, (short)0, acc[nb], false, false);
        }
    }
    #pragma unroll
    for (int nb = 0; nb < 4; ++nb) {
        const int ncol = n0 + nb * 16 + ln;
        const int head = ncol >> 6;
        const int d    = ncol & 63;
        const float bb = bias[ncol];
        #pragma unroll
        for (int r = 0; r < 8; ++r) {
            const int m = m0 + r + hi;
            const int b = m >> 10;
            const int s = m & 1023;
            size_t idx;
            if (z == 2)   // V stored d-major per head: (b,h,d,s)
                idx = (((size_t)(b * NH_ + head)) * D_ + d) * S_LEN + s;
            else          // Q,K stored s-major per head: (b,h,s,d)
                idx = (((size_t)(b * NH_ + head)) * S_LEN + s) * D_ + d;
            dst[idx] = f2bf(acc[nb][r] + bb);
        }
    }
}

// ---------------------------------------------------------------------------
// Kernel 2: Qrel[b,h,q,r] = sum_d Q[b,h,q,d] * rel_emb[r,d]   (41 r values)
// ---------------------------------------------------------------------------
__global__ void qrel_kernel(const unsigned short* __restrict__ Qh,
                            const float* __restrict__ rel_emb,
                            float* __restrict__ Qrel)
{
    const int idx   = blockIdx.x * blockDim.x + threadIdx.x;
    const int total = N_B * NH_ * S_LEN * NREL;
    if (idx >= total) return;
    const int r = idx % NREL;
    const int t = idx / NREL;             // (b*NH+h)*S + q
    uint4 qv[8];
    #pragma unroll
    for (int i = 0; i < 8; ++i) qv[i] = ((const uint4*)(Qh + (size_t)t * D_))[i];
    const unsigned short* qu = (const unsigned short*)qv;
    const float* re = rel_emb + r * D_;
    float s = 0.0f;
    #pragma unroll 8
    for (int d = 0; d < D_; ++d) s += bf2f(qu[d]) * re[d];
    Qrel[(size_t)t * QRS + r] = s;
}

// ---------------------------------------------------------------------------
// Kernel 3: flash attention with online softmax.
// grid = (S/64, NH, N), block = 128. Each wave: 16 queries, streams 32-key tiles.
// ---------------------------------------------------------------------------
__global__ __launch_bounds__(128) void attn_kernel(
    const unsigned short* __restrict__ Qh, const unsigned short* __restrict__ Kh,
    const unsigned short* __restrict__ Vt, const float* __restrict__ Qrel,
    const unsigned char* __restrict__ mask,
    float* __restrict__ ctx_f32, unsigned short* __restrict__ ctx_bf)
{
    __shared__ __align__(16) unsigned short Pst[4][16][32];   // per-wave P staging

    const int lane  = threadIdx.x & 31;
    const int w     = threadIdx.x >> 5;
    const int ln    = lane & 15;
    const int hi    = (lane & 16) ? 8 : 0;
    const int koffA = (lane & 16) ? 16 : 0;
    const int koffB = (lane & 16) ? 16 : 0;
    const int b     = blockIdx.z;
    const int h     = blockIdx.y;
    const int qbase = blockIdx.x * 64 + w * 16;

    const size_t headoff = ((size_t)(b * NH_ + h)) * S_LEN;
    const unsigned short* Qp = Qh + headoff * D_;
    const unsigned short* Kp = Kh + headoff * D_;
    const unsigned short* Vp = Vt + headoff * D_;   // (b,h,d,s): row d has S keys
    const float* Qr = Qrel + headoff * QRS;

    // Q A-fragments, held for the whole key loop
    BF16Frag aq[2];
    #pragma unroll
    for (int kh = 0; kh < 2; ++kh) {
        const unsigned short* p = Qp + (size_t)(qbase + ln) * D_ + kh * 32 + koffA;
        aq[kh].q[0] = *(const uint4*)(p);
        aq[kh].q[1] = *(const uint4*)(p + 8);
    }

    v8f o[4];
    #pragma unroll
    for (int i = 0; i < 4; ++i) o[i] = v8f_zero();
    float mrow[8], lrow[8];
    const float NEG = -__builtin_inff();
    #pragma unroll
    for (int r = 0; r < 8; ++r) { mrow[r] = NEG; lrow[r] = 0.0f; }

    const int nkt = ((qbase + 15) >> 5) + 1;   // causal bound on key tiles
    for (int kt = 0; kt < nkt; ++kt) {
        const int kb = kt * 32;

        // speculative prefetch of the next key tile: 32 K-rows (one line each,
        // spread over lanes) and all 64 V-rows at the next key offset
        const int kbn = kb + 32;
        __builtin_prefetch(Kp + (size_t)(kbn + lane) * D_, 0, 3);
        __builtin_prefetch(Vp + (size_t)lane * S_LEN + kbn, 0, 3);
        __builtin_prefetch(Vp + (size_t)(lane + 32) * S_LEN + kbn, 0, 3);

        // ---- QK^T for 32 keys -> two 16x16 f32 score tiles
        v8f sc[2];
        #pragma unroll
        for (int nb = 0; nb < 2; ++nb) {
            v8f c = v8f_zero();
            const int key = kb + nb * 16 + ln;
            #pragma unroll
            for (int kh = 0; kh < 2; ++kh) {
                BF16Frag bkf;
                const unsigned short* p = Kp + (size_t)key * D_ + kh * 32 + koffB;
                bkf.q[0] = *(const uint4*)(p);
                bkf.q[1] = *(const uint4*)(p + 8);
                c = __builtin_amdgcn_wmma_f32_16x16x32_bf16(
                    false, aq[kh].v, false, bkf.v, (short)0, c, false, false);
            }
            sc[nb] = c;
        }

        // ---- bias + mask + online softmax (rows live in 16-lane halves)
        #pragma unroll
        for (int r = 0; r < 8; ++r) {
            const int m = r + hi;
            const int q = qbase + m;
            float e01[2];
            #pragma unroll
            for (int nb = 0; nb < 2; ++nb) {
                const int key = kb + nb * 16 + ln;
                int rr = q - key;
                rr = rr < -P_ ? -P_ : (rr > P_ ? P_ : rr);
                float en = sc[nb][r] * (1.0f / 64.0f)
                         + Qr[(size_t)q * QRS + (rr + P_)] * 0.125f;
                const bool ok = (key <= q) && (mask[b * S_LEN + key] != 0);
                e01[nb] = ok ? en : NEG;
            }
            float rmax = fmaxf(e01[0], e01[1]);
            #pragma unroll
            for (int off = 1; off < 16; off <<= 1)
                rmax = fmaxf(rmax, __shfl_xor(rmax, off, 32));
            const float newm = fmaxf(mrow[r], rmax);
            const float corr = __expf(mrow[r] - newm);
            const float p0 = __expf(e01[0] - newm);
            const float p1 = __expf(e01[1] - newm);
            float ps = p0 + p1;
            #pragma unroll
            for (int off = 1; off < 16; off <<= 1)
                ps += __shfl_xor(ps, off, 32);
            lrow[r] = lrow[r] * corr + ps;
            mrow[r] = newm;
            #pragma unroll
            for (int nb2 = 0; nb2 < 4; ++nb2) o[nb2][r] *= corr;
            Pst[w][m][ln]      = f2bf(p0);
            Pst[w][m][16 + ln] = f2bf(p1);
        }

        // ---- restage P (16x32) into A-fragment layout (2 x ds_load_b128)
        BF16Frag ap;
        ap.q[0] = *(const uint4*)&Pst[w][ln][koffA];
        ap.q[1] = *(const uint4*)&Pst[w][ln][koffA + 8];

        // ---- P @ V   (V is d-major: row dcol holds S contiguous keys)
        #pragma unroll
        for (int nb2 = 0; nb2 < 4; ++nb2) {
            const int dcol = nb2 * 16 + ln;
            BF16Frag bvf;
            const unsigned short* p = Vp + (size_t)dcol * S_LEN + kb + koffB;
            bvf.q[0] = *(const uint4*)(p);
            bvf.q[1] = *(const uint4*)(p + 8);
            o[nb2] = __builtin_amdgcn_wmma_f32_16x16x32_bf16(
                false, ap.v, false, bvf.v, (short)0, o[nb2], false, false);
        }
    }

    // ---- epilogue: normalize and write ctx (f32 to d_out, bf16 to workspace)
    #pragma unroll
    for (int r = 0; r < 8; ++r) {
        const int m = r + hi;
        const int q = qbase + m;
        const float inv = 1.0f / lrow[r];
        #pragma unroll
        for (int nb2 = 0; nb2 < 4; ++nb2) {
            const int col = h * D_ + nb2 * 16 + ln;
            const float val = o[nb2][r] * inv;
            const size_t idx = ((size_t)(b * S_LEN + q)) * H_DIM + col;
            ctx_f32[idx] = val;
            ctx_bf[idx]  = f2bf(val);
        }
    }
}

// ---------------------------------------------------------------------------
// Kernel 4: out = ctx @ Wo + bo   (bf16 WMMA, f32 out; Wo pre-transposed bf16)
// ---------------------------------------------------------------------------
__global__ __launch_bounds__(128) void oproj_kernel(
    const unsigned short* __restrict__ ctx_bf, const unsigned short* __restrict__ Wot,
    const float* __restrict__ bo, float* __restrict__ out)
{
    const int lane  = threadIdx.x & 31;
    const int w     = threadIdx.x >> 5;
    const int ln    = lane & 15;
    const int hi    = (lane & 16) ? 8 : 0;
    const int koffA = (lane & 16) ? 16 : 0;
    const int koffB = (lane & 16) ? 16 : 0;
    const int m0    = blockIdx.x * 64 + w * 16;
    const int n0    = blockIdx.y * 64;

    v8f acc[4];
    #pragma unroll
    for (int i = 0; i < 4; ++i) acc[i] = v8f_zero();

    const unsigned short* Arow = ctx_bf + (size_t)(m0 + ln) * H_DIM;
    const unsigned short* Brow[4];
    #pragma unroll
    for (int nb = 0; nb < 4; ++nb)
        Brow[nb] = Wot + (size_t)(n0 + nb * 16 + ln) * H_DIM;

    for (int kt = 0; kt < H_DIM / 32; ++kt) {
        const int k0 = kt * 32;
        const int kpf = k0 + 64;
        __builtin_prefetch(Arow + kpf, 0, 3);
        __builtin_prefetch(Brow[0] + kpf, 0, 3);
        __builtin_prefetch(Brow[1] + kpf, 0, 3);
        __builtin_prefetch(Brow[2] + kpf, 0, 3);
        __builtin_prefetch(Brow[3] + kpf, 0, 3);

        BF16Frag a;
        a.q[0] = *(const uint4*)(Arow + k0 + koffA);
        a.q[1] = *(const uint4*)(Arow + k0 + koffA + 8);
        #pragma unroll
        for (int nb = 0; nb < 4; ++nb) {
            BF16Frag bw;
            bw.q[0] = *(const uint4*)(Brow[nb] + k0 + koffB);
            bw.q[1] = *(const uint4*)(Brow[nb] + k0 + koffB + 8);
            acc[nb] = __builtin_amdgcn_wmma_f32_16x16x32_bf16(
                false, a.v, false, bw.v, (short)0, acc[nb], false, false);
        }
    }
    #pragma unroll
    for (int nb = 0; nb < 4; ++nb) {
        const int ncol = n0 + nb * 16 + ln;
        const float bb = bo[ncol];
        #pragma unroll
        for (int r = 0; r < 8; ++r) {
            const int m = m0 + r + hi;
            out[(size_t)m * H_DIM + ncol] = acc[nb][r] + bb;
        }
    }
}

// ---------------------------------------------------------------------------
extern "C" void kernel_launch(void* const* d_in, const int* in_sizes, int n_in,
                              void* d_out, int out_size, void* d_ws, size_t ws_size,
                              hipStream_t stream)
{
    const float* query = (const float*)d_in[0];
    const float* key   = (const float*)d_in[1];
    const float* value = (const float*)d_in[2];
    const unsigned char* mask = (const unsigned char*)d_in[3];   // jax bool = 1 byte
    const float* Wq = (const float*)d_in[4];
    const float* bq = (const float*)d_in[5];
    const float* Wk = (const float*)d_in[6];
    const float* bk = (const float*)d_in[7];
    const float* Wv = (const float*)d_in[8];
    const float* bv = (const float*)d_in[9];
    const float* Wo = (const float*)d_in[10];
    const float* bo = (const float*)d_in[11];
    const float* rel_emb = (const float*)d_in[12];

    float* out_ptr = (float*)d_out;                                 // (N,S,H)
    float* ctx_ptr = out_ptr + (size_t)N_B * S_LEN * H_DIM;         // (N,S,H)

    // workspace layout (~77 MB, arrays packed back-to-back)
    const size_t NE = (size_t)N_B * S_LEN * H_DIM;                  // 4,194,304
    const size_t WE = (size_t)H_DIM * H_DIM;                        // 1,048,576
    unsigned short* Xqb  = (unsigned short*)d_ws;
    unsigned short* Xkb  = Xqb + NE;
    unsigned short* Xvb  = Xkb + NE;
    unsigned short* Wqt  = Xvb + NE;
    unsigned short* Wkt  = Wqt + WE;
    unsigned short* Wvt  = Wkt + WE;
    unsigned short* Wot  = Wvt + WE;
    unsigned short* Qh   = Wot + WE;
    unsigned short* Kh   = Qh + NE;
    unsigned short* Vt   = Kh + NE;
    unsigned short* Ctxb = Vt + NE;
    float*          Qrel = (float*)(Ctxb + NE);                     // N*NH*S*QRS floats

    const dim3 blk(128, 1, 1);

    // 0) pack inputs to bf16; weights to transposed bf16
    {
        const int n4 = (int)(NE / 4);
        pack_bf16_kernel<<<(n4 + 255) / 256, 256, 0, stream>>>(query, Xqb, n4);
        pack_bf16_kernel<<<(n4 + 255) / 256, 256, 0, stream>>>(key,   Xkb, n4);
        pack_bf16_kernel<<<(n4 + 255) / 256, 256, 0, stream>>>(value, Xvb, n4);
        const int nw = (int)WE;
        pack_wT_kernel<<<(nw + 255) / 256, 256, 0, stream>>>(Wq, Wqt);
        pack_wT_kernel<<<(nw + 255) / 256, 256, 0, stream>>>(Wk, Wkt);
        pack_wT_kernel<<<(nw + 255) / 256, 256, 0, stream>>>(Wv, Wvt);
        pack_wT_kernel<<<(nw + 255) / 256, 256, 0, stream>>>(Wo, Wot);
    }

    // 1) QKV projections (3 GEMMs via grid.z); V written d-major
    qkv_proj_kernel<<<dim3((N_B * S_LEN) / 64, H_DIM / 64, 3), blk, 0, stream>>>(
        Xqb, Xkb, Xvb, Wqt, Wkt, Wvt, bq, bk, bv, Qh, Kh, Vt);

    // 2) relative-position dot table
    {
        const int total = N_B * NH_ * S_LEN * NREL;
        qrel_kernel<<<(total + 255) / 256, 256, 0, stream>>>(Qh, rel_emb, Qrel);
    }

    // 3) flash attention -> ctx (f32 into d_out, bf16 into ws)
    attn_kernel<<<dim3(S_LEN / 64, NH_, N_B), blk, 0, stream>>>(
        Qh, Kh, Vt, Qrel, mask, ctx_ptr, Ctxb);

    // 4) output projection
    oproj_kernel<<<dim3((N_B * S_LEN) / 64, H_DIM / 64, 1), blk, 0, stream>>>(
        Ctxb, Wot, bo, out_ptr);
}
